// FourierKernelAttention_1236950581945
// MI455X (gfx1250) — matrix operand order repaired and checked by reading, not compiled, not run on hardware
//
#include <hip/hip_runtime.h>

// Fourier kernel (linear) attention, CDNA5 / gfx1250, wave32 + WMMA f16->f32.
//   B=4, SK=SQ=4096, H=16, D=64, N=32  -> per-(h,b): KV = Kfeat^T V (64x64),
//   Out = Qfeat KV. The reference's 4N-wide feature map collapses to 2N=64.
// Workspace: 8 MB KV f32 partials + 512 KB reduced f16 KV (transposed).

static constexpr int kH   = 16;
static constexpr int kD   = 64;
static constexpr int kHD  = kH * kD;         // 1024
static constexpr int kSK  = 4096;
static constexpr int kSQ  = 4096;
static constexpr int kChunks = 8;
static constexpr int kKPC = kSK / kChunks;   // keys per chunk = 512
static constexpr float kFreqScale = 3.2429343520926866f;   // 32*pi/31
static constexpr float kHalfPi    = 1.5707963267948966f;

typedef __attribute__((ext_vector_type(16))) _Float16 v16h;
typedef __attribute__((ext_vector_type(8)))  _Float16 v8h;
typedef __attribute__((ext_vector_type(2)))  _Float16 v2h;
typedef __attribute__((ext_vector_type(8)))  float    v8f;
typedef __attribute__((ext_vector_type(4)))  float    v4f;

// A-fragment (16x32 f16, M x K): lane holds row M=lane&15; elements 0..7 are
// K = kb..kb+7, elements 8..15 are K = 16+kb..16+kb+7, kb=(lane>=16)?8:0.
__device__ __forceinline__ v16h load_a_frag(const _Float16* p) {
  v8h lo = *reinterpret_cast<const v8h*>(p);
  v8h hi = *reinterpret_cast<const v8h*>(p + 16);
  v16h r;
#pragma unroll
  for (int i = 0; i < 8; ++i) { r[i] = lo[i]; r[i + 8] = hi[i]; }
  return r;
}

// B-fragment (32x16 f16, K x N): lane holds column N=lane&15; elements 0..7 are
// K = kb..kb+7, elements 8..15 are K = 16+kb..16+kb+7 (B kept transposed).
__device__ __forceinline__ v16h load_b_frag(const _Float16* p) {
  v8h lo = *reinterpret_cast<const v8h*>(p);
  v8h hi = *reinterpret_cast<const v8h*>(p + 16);
  v16h r;
#pragma unroll
  for (int i = 0; i < 8; ++i) { r[i] = lo[i]; r[i + 8] = hi[i]; }
  return r;
}

// ------------- Phase 1: KV partials = Kfeat^T (64 x 512) * V (512 x 64) -----
__global__ __launch_bounds__(256) void fka_phase1(
    const float* __restrict__ key_in,   // (B, SK, H)
    const float* __restrict__ value,    // (B, SK, H*D)
    float* __restrict__ part) {         // (chunks, 64hb, 64f, 64d)
  const int hb    = blockIdx.x;         // 0..63
  const int chunk = blockIdx.y;         // 0..7
  const int h = hb >> 2, b = hb & 3;
  const int tid  = threadIdx.x;
  const int wave = tid >> 5;
  const int lane = tid & 31;
  const int mrow = lane & 15;
  const int kb   = (lane >> 4) * 8;
  const int kb2  = (lane >> 4) * 16;

  __shared__ alignas(16) _Float16 Als[2][64][40];  // [buf][feature][key]
  __shared__ alignas(16) _Float16 Bt [2][64][40];  // [buf][d][key]
  __shared__ float kall[kKPC];

  const int kbase = chunk * kKPC;
#pragma unroll
  for (int r = 0; r < kKPC / 256; ++r) {
    int kk = tid + r * 256;
    kall[kk] = key_in[((size_t)b * kSK + kbase + kk) * kH + h];
  }
  __syncthreads();

  // value staging: thread owns keys {2kp, 2kp+1} x d0..d0+3 (two b128 loads)
  const int d0 = (tid & 15) * 4;
  const int kp = tid >> 4;                 // 0..15
  const float* vptr = value + ((size_t)b * kSK + kbase + 2 * kp) * kHD + h * kD + d0;
  v4f r0 = *reinterpret_cast<const v4f*>(vptr);
  v4f r1 = *reinterpret_cast<const v4f*>(vptr + kHD);

  // trig staging: slot pi = tid + r*256 -> key-pair kp2 = tid&15 (r-invariant),
  // feature row f = (tid>>4) + 16r, n = f&31, and f<32 <=> r<2 (compile time).
  const int kp2   = tid & 15;
  const int fbase = tid >> 4;              // 0..15
  const float nsc0 = kFreqScale * (float)(fbase);        // rows fbase, fbase+32
  const float nsc1 = kFreqScale * (float)(fbase + 16);   // rows fbase+16, +48

  const int tile0 = wave * 2;              // 16 output tiles over 8 waves
  const int mt  = tile0 >> 2;
  const int nt0 = tile0 & 3, nt1 = (tile0 + 1) & 3;
  v8f acc0 = {}; v8f acc1 = {};

#pragma unroll 2
  for (int it = 0; it < kKPC / 32; ++it) {
    const int cur = it & 1;
    // store value tile (packed key-pairs -> b32 stores)
#pragma unroll
    for (int i = 0; i < 4; ++i) {
      v2h p = { (_Float16)r0[i], (_Float16)r1[i] };
      *reinterpret_cast<v2h*>(&Bt[cur][d0 + i][2 * kp]) = p;
    }
    // trig features: cos rows use sin(x + pi/2) so there is no branch at all
    {
      float k0 = kall[it * 32 + 2 * kp2];
      float k1 = kall[it * 32 + 2 * kp2 + 1];
#pragma unroll
      for (int r = 0; r < 4; ++r) {
        const int   f    = fbase + 16 * r;
        const float nsc  = (r & 1) ? nsc1 : nsc0;
        const float offs = (r < 2) ? kHalfPi : 0.0f;   // cos half / sin half
        float t0 = __sinf(nsc * k0 + offs);
        float t1 = __sinf(nsc * k1 + offs);
        v2h p = { (_Float16)t0, (_Float16)t1 };
        *reinterpret_cast<v2h*>(&Als[cur][f][2 * kp2]) = p;
      }
    }
    __syncthreads();
    // prefetch next value tile while WMMAs run
    if (it < kKPC / 32 - 1) {
      vptr += 32 * kHD;
      r0 = *reinterpret_cast<const v4f*>(vptr);
      r1 = *reinterpret_cast<const v4f*>(vptr + kHD);
    }
    v16h a  = load_a_frag(&Als[cur][mt * 16 + mrow][kb]);
    v16h b0 = load_b_frag(&Bt[cur][nt0 * 16 + mrow][kb]);
    v16h b1 = load_b_frag(&Bt[cur][nt1 * 16 + mrow][kb]);
    acc0 = __builtin_amdgcn_wmma_f32_16x16x32_f16(false, a, false, b0, (short)0, acc0, false, false);
    acc1 = __builtin_amdgcn_wmma_f32_16x16x32_f16(false, a, false, b1, (short)0, acc1, false, false);
    // one barrier/iter: buf[x] readers (iter it) pass the barrier in iter it+1
    // before buf[x] is overwritten in iter it+2.
  }

  // C/D layout: element j -> row M = j + (lane>=16 ? 8 : 0), col N = lane&15
  float* dst = part + (((size_t)chunk * 64 + hb) * 64) * 64;
#pragma unroll
  for (int j = 0; j < 8; ++j) {
    int f = mt * 16 + j + (lane >> 4) * 8;
    dst[f * 64 + nt0 * 16 + mrow] = acc0[j];
    dst[f * 64 + nt1 * 16 + mrow] = acc1[j];
  }
}

// ------------- Phase 1.5: reduce partials -> f16 KV, transposed [d][f] ------
__global__ __launch_bounds__(256) void fka_reduce(
    const float* __restrict__ part,     // (chunks, 64hb, 64f, 64d)
    _Float16* __restrict__ kvred) {     // (64hb, 64d, 64f)
  const int hb = blockIdx.x;
  const int tid = threadIdx.x;
  __shared__ alignas(16) _Float16 T[64][72];

#pragma unroll
  for (int r = 0; r < 4; ++r) {
    int sl = tid + r * 256;               // 1024 slots: (f, d-quad)
    int f = sl >> 4, d4 = (sl & 15) * 4;
    v4f s = {};
#pragma unroll
    for (int c = 0; c < kChunks; ++c)
      s += *reinterpret_cast<const v4f*>(&part[(((size_t)c * 64 + hb) * 64 + f) * 64 + d4]);
#pragma unroll
    for (int i = 0; i < 4; ++i) T[d4 + i][f] = (_Float16)s[i];
  }
  __syncthreads();
#pragma unroll
  for (int r = 0; r < 2; ++r) {
    int t2 = tid + r * 256;               // 512 b128 slots
    int d = t2 >> 3, f0 = (t2 & 7) * 8;
    *reinterpret_cast<v8h*>(kvred + (size_t)hb * 4096 + d * 64 + f0) =
        *reinterpret_cast<const v8h*>(&T[d][f0]);
  }
}

// ------------- Phase 2: Out(64q x 64d) = Qfeat(64x64) * KV(64x64) -----------
__global__ __launch_bounds__(256) void fka_phase2(
    const float* __restrict__ query,    // (B, SQ, H)
    const float* __restrict__ ramps,    // (H, 1, 1, N)
    const float* __restrict__ iamps,    // (H, 1, 1, N)
    const _Float16* __restrict__ kvred, // (64hb, 64d, 64f)
    float* __restrict__ out) {          // (B, SQ, H*D)
  const int hb  = blockIdx.x;           // 0..63
  const int sq0 = blockIdx.y * 64;
  const int h = hb >> 2, b = hb & 3;
  const int tid  = threadIdx.x;
  const int wave = tid >> 5;
  const int lane = tid & 31;
  const int mrow = lane & 15;
  const int kb   = (lane >> 4) * 8;
  const int kb2  = (lane >> 4) * 16;

  __shared__ alignas(16) _Float16 A2 [64][72];  // [query][feature]
  __shared__ alignas(16) _Float16 KVt[64][72];  // [d][feature]
  __shared__ float qv[64];
  __shared__ float ramp_s[32];
  __shared__ float iamp_s[32];

  if (tid < 64)        qv[tid] = query[((size_t)b * kSQ + sq0 + tid) * kH + h];
  else if (tid < 96)   ramp_s[tid - 64] = ramps[h * 32 + (tid - 64)];
  else if (tid < 128)  iamp_s[tid - 96] = iamps[h * 32 + (tid - 96)];

  // stage reduced KV (already f16 + transposed): pure b128 copies
#pragma unroll
  for (int r = 0; r < 2; ++r) {
    int t2 = tid + r * 256;
    int d = t2 >> 3, f0 = (t2 & 7) * 8;
    *reinterpret_cast<v8h*>(&KVt[d][f0]) =
        *reinterpret_cast<const v8h*>(kvred + (size_t)hb * 4096 + d * 64 + f0);
  }
  __syncthreads();

  // query features, one sincos per (q, n):
  //   A2[q][n]    = re_n*cos + im_n*sin ;  A2[q][32+n] = re_n*sin - im_n*cos
#pragma unroll
  for (int r = 0; r < 8; ++r) {
    int li = tid + r * 256;               // 2048 (q, n) slots
    int q = li >> 5, n = li & 31;
    float ph = kFreqScale * (float)n * qv[q];
    float sn, cs;
    __sincosf(ph, &sn, &cs);
    A2[q][n]      = (_Float16)(ramp_s[n] * cs + iamp_s[n] * sn);
    A2[q][32 + n] = (_Float16)(ramp_s[n] * sn - iamp_s[n] * cs);
  }
  __syncthreads();

  const int tile0 = wave * 2;
  const int qt  = tile0 >> 2;
  const int dt0 = tile0 & 3, dt1 = (tile0 + 1) & 3;
  v8f acc0 = {}; v8f acc1 = {};
#pragma unroll
  for (int ks = 0; ks < 64; ks += 32) {
    v16h a  = load_a_frag(&A2[qt * 16 + mrow][ks + kb]);
    v16h b0 = load_b_frag(&KVt[dt0 * 16 + mrow][ks + kb]);
    v16h b1 = load_b_frag(&KVt[dt1 * 16 + mrow][ks + kb]);
    acc0 = __builtin_amdgcn_wmma_f32_16x16x32_f16(false, a, false, b0, (short)0, acc0, false, false);
    acc1 = __builtin_amdgcn_wmma_f32_16x16x32_f16(false, a, false, b1, (short)0, acc1, false, false);
  }
#pragma unroll
  for (int j = 0; j < 8; ++j) {
    int q = qt * 16 + j + (lane >> 4) * 8;
    size_t row = ((size_t)b * kSQ + sq0 + q) * kHD + (size_t)h * kD;
    out[row + dt0 * 16 + mrow] = acc0[j];
    out[row + dt1 * 16 + mrow] = acc1[j];
  }
}

extern "C" void kernel_launch(void* const* d_in, const int* in_sizes, int n_in,
                              void* d_out, int out_size, void* d_ws, size_t ws_size,
                              hipStream_t stream) {
  const float* key_in = (const float*)d_in[0];
  const float* value  = (const float*)d_in[1];
  const float* query  = (const float*)d_in[2];
  const float* ramps  = (const float*)d_in[3];
  const float* iamps  = (const float*)d_in[4];
  float* out  = (float*)d_out;

  float*    part  = (float*)d_ws;                                   // 8 MB
  _Float16* kvred = (_Float16*)((char*)d_ws + (size_t)kChunks * 64 * 64 * 64 * 4);

  fka_phase1<<<dim3(64, kChunks), 256, 0, stream>>>(key_in, value, part);
  fka_reduce<<<64, 256, 0, stream>>>(part, kvred);
  fka_phase2<<<dim3(64, kSQ / 64), 256, 0, stream>>>(query, ramps, iamps, kvred, out);

  (void)in_sizes; (void)n_in; (void)out_size; (void)ws_size;
}
